// NTXent_54812372631900
// MI455X (gfx1250) — matrix operand order repaired and checked by reading, not compiled, not run on hardware
//
#include <hip/hip_runtime.h>
#include <stdint.h>

// ---------------- problem constants ----------------
#define B_N   8192          // batch (rows of S, cols of S)
#define D_K   626           // feature dim
#define KP    648           // padded row stride in elements (>=640 used, banks: 1296B = 324 words = 4 mod 64)
#define KC    20            // K chunks of 32 -> covers 640 >= 626 (tail zero-padded)
#define INV_T 4.0f          // 1 / 0.25
#define MBLK  64            // rows per workgroup
#define NSTG  32            // columns staged per LDS stage
#define NSTAGES (B_N / NSTG)

#define LDS_MAT  (NSTG * KP * 2)    // 41472 B: one 32-row bf16 panel
#define LDS_BUF  (2 * LDS_MAT)      // hi + lo = 82944 B
#define LDS_TOT  (2 * LDS_BUF)      // double buffered = 165888 B (< 320 KB/WGP)

typedef __attribute__((ext_vector_type(16))) __bf16 v16bf;
typedef __attribute__((ext_vector_type(8)))  float  v8f;
typedef __attribute__((ext_vector_type(4)))  unsigned int v4u;
typedef __attribute__((ext_vector_type(8)))  int v8i;
typedef __attribute__((ext_vector_type(4)))  int v4i;

union Frag { v16bf v; uint4 q[2]; };

#if defined(__has_builtin)
# if __has_builtin(__builtin_amdgcn_tensor_load_to_lds)
#  define HAS_TDM 1
# endif
#endif
#ifndef HAS_TDM
# define HAS_TDM 0
#endif

// ---------------- phase 1: normalize + bf16 hi/lo split ----------------
__global__ __launch_bounds__(256) void prep_kernel(const float* __restrict__ anchors,
                                                   const float* __restrict__ poss,
                                                   __bf16* __restrict__ Ahi, __bf16* __restrict__ Alo,
                                                   __bf16* __restrict__ Phi, __bf16* __restrict__ Plo) {
  const int wave = threadIdx.x >> 5, lane = threadIdx.x & 31;
  const int r = blockIdx.x * 8 + wave;            // 0 .. 2*B_N-1
  const float* src; __bf16* dh; __bf16* dl; int row;
  if (r < B_N) { src = anchors + (size_t)r * D_K; dh = Ahi; dl = Alo; row = r; }
  else         { src = poss + (size_t)(r - B_N) * D_K; dh = Phi; dl = Plo; row = r - B_N; }

  float ssq = 0.0f;
  for (int k = lane; k < D_K; k += 32) { float x = src[k]; ssq += x * x; }
  #pragma unroll
  for (int off = 16; off; off >>= 1) ssq += __shfl_xor(ssq, off, 32);
  const float inv = 1.0f / fmaxf(sqrtf(ssq), 1e-12f);   // norms ~25, eps clamp never binds

  dh += (size_t)row * KP; dl += (size_t)row * KP;
  for (int k = lane; k < KP; k += 32) {
    float x = (k < D_K) ? src[k] * inv : 0.0f;
    __bf16 h = (__bf16)x;
    __bf16 l = (__bf16)(x - (float)h);
    dh[k] = h; dl[k] = l;
  }
}

// ---------------- TDM issue: copy one 32-row x 648-elem bf16 panel to LDS ----------------
__device__ __forceinline__ void tdm_issue(unsigned lds_off, const __bf16* gptr) {
#if HAS_TDM
  unsigned long long ga = (unsigned long long)(uintptr_t)gptr;
  v4u g0;
  g0[0] = 1u;                                               // count=1, no gather, user D#
  g0[1] = lds_off;                                          // LDS byte address
  g0[2] = (unsigned)(ga & 0xFFFFFFFFu);                     // global addr lo
  g0[3] = (unsigned)((ga >> 32) & 0x01FFFFFFu) | 0x80000000u; // addr hi | type=2 (bits 127:126 = 2)
  v8i g1;
  g1[0] = 0x00010000;                  // wg_mask=0, data_size=1 (2 bytes), no flags
  g1[1] = (int)((unsigned)KP << 16);   // [47:32] atomic_bar=0, [63:48] tensor_dim0 lo16 = 648
  g1[2] = (int)((unsigned)B_N << 16);  // [79:64] tensor_dim0 hi16=0, [95:80] tensor_dim1 lo16 = 8192
  g1[3] = (int)((unsigned)KP << 16);   // [111:96] tensor_dim1 hi16=0, [127:112] tile_dim0 = 648
  g1[4] = NSTG;                        // [143:128] tile_dim1 = 32, [159:144] tile_dim2 = 0
  g1[5] = KP;                          // tensor_dim0_stride lo32 = 648
  g1[6] = 0;                           // stride0 hi16 = 0, stride1 lo16 = 0
  g1[7] = 0;
  v4i z4 = {};
  v8i z8 = {};
  __builtin_amdgcn_tensor_load_to_lds(g0, g1, z4, z4, z8, 0);
#else
  (void)lds_off; (void)gptr;
#endif
}

// ---------------- phase 2: fused GEMM (bf16x3 WMMA) + online LSE / rowsum / diag ----------------
__global__ __launch_bounds__(128) void ntxent_main(const __bf16* __restrict__ Ahi,
                                                   const __bf16* __restrict__ Alo,
                                                   const __bf16* __restrict__ Phi,
                                                   const __bf16* __restrict__ Plo,
                                                   float* __restrict__ loss_row,
                                                   float* __restrict__ diag_row,
                                                   float* __restrict__ rowsum_row) {
  extern __shared__ char smem[];
  const int tid  = threadIdx.x;
  const int wave = tid >> 5, lane = tid & 31;
  const int half = lane >> 4, lr = lane & 15;
  const int m0w  = blockIdx.x * MBLK + wave * 16;   // this wave's 16 rows
  const int rowA = m0w + lr;
  const int rowbase = m0w + half * 8;               // rows carried in this lane's accumulators

  const __bf16* arh = Ahi + (size_t)rowA * KP;      // A fragments served from L2 (42MB < 192MB)
  const __bf16* arl = Alo + (size_t)rowA * KP;

  const unsigned lds_base = (unsigned)__builtin_amdgcn_groupstaticsize();

  float mx[8], sm[8], rs[8], dg[8];
  #pragma unroll
  for (int r = 0; r < 8; ++r) { mx[r] = -1e30f; sm[r] = 0.0f; rs[r] = 0.0f; dg[r] = 0.0f; }

#if HAS_TDM
  if (wave == 0) {                                  // prologue: stage 0 -> buf 0 (hi, lo)
    tdm_issue(lds_base,            Phi);
    tdm_issue(lds_base + LDS_MAT,  Plo);
  }
#endif

  for (int s = 0; s < NSTAGES; ++s) {
    const int buf = s & 1;
#if HAS_TDM
    if (wave == 0) {
      if (s + 1 < NSTAGES) {                        // prefetch next stage into other buffer
        const unsigned nb = (unsigned)(((s + 1) & 1) * LDS_BUF);
        tdm_issue(lds_base + nb,           Phi + (size_t)(s + 1) * NSTG * KP);
        tdm_issue(lds_base + nb + LDS_MAT, Plo + (size_t)(s + 1) * NSTG * KP);
        __builtin_amdgcn_s_wait_tensorcnt(2);       // 2 newest outstanding -> stage s landed
      } else {
        __builtin_amdgcn_s_wait_tensorcnt(0);
      }
    }
    __syncthreads();                                // publish staged panel to all waves
#else
    {                                               // cooperative fallback copy
      const uint4* gh = (const uint4*)(Phi + (size_t)s * NSTG * KP);
      const uint4* gl = (const uint4*)(Plo + (size_t)s * NSTG * KP);
      uint4* sh = (uint4*)(smem + buf * LDS_BUF);
      uint4* sl = (uint4*)(smem + buf * LDS_BUF + LDS_MAT);
      for (int i = tid; i < (NSTG * KP * 2) / 16; i += 128) { sh[i] = gh[i]; sl[i] = gl[i]; }
      __syncthreads();
    }
#endif

    const int n0 = s * NSTG;
    const char* bufp = smem + buf * LDS_BUF;
    v8f acc0 = {}, acc1 = {};

    #pragma unroll
    for (int c = 0; c < KC; ++c) {
      // A 16x32 bf16 layout: lane<16 holds K {0..7,16..23}, lane>=16 holds {8..15,24..31}
      const int kb = c * 32 + half * 8;
      Frag ah, al;
      ah.q[0] = *(const uint4*)(arh + kb);
      ah.q[1] = *(const uint4*)(arh + kb + 16);
      al.q[0] = *(const uint4*)(arl + kb);
      al.q[1] = *(const uint4*)(arl + kb + 16);
      #pragma unroll
      for (int nt = 0; nt < 2; ++nt) {
        // B 32x16: lane<16 col=lr K {0..7,16..23}; lane>=16 col=lr K {8..15,24..31}
        const int col = nt * 16 + lr;
        const int ko  = c * 32 + half * 8;
        const char* bp = bufp + ((size_t)col * KP + ko) * 2;
        Frag bh, bl;
        bh.q[0] = *(const uint4*)(bp);
        bh.q[1] = *(const uint4*)(bp + 32);
        bl.q[0] = *(const uint4*)(bp + LDS_MAT);
        bl.q[1] = *(const uint4*)(bp + LDS_MAT + 32);
        v8f& acc = nt ? acc1 : acc0;
        acc = __builtin_amdgcn_wmma_f32_16x16x32_bf16(false, ah.v, false, bh.v, (short)0, acc, false, false);
        acc = __builtin_amdgcn_wmma_f32_16x16x32_bf16(false, ah.v, false, bl.v, (short)0, acc, false, false);
        acc = __builtin_amdgcn_wmma_f32_16x16x32_bf16(false, al.v, false, bh.v, (short)0, acc, false, false);
      }
    }

    // epilogue: acc[r] = S[rowbase+r][n], n = this lane's column
    #pragma unroll
    for (int nt = 0; nt < 2; ++nt) {
      const v8f& acc = nt ? acc1 : acc0;
      const int n = n0 + nt * 16 + lr;
      #pragma unroll
      for (int r = 0; r < 8; ++r) {
        const float v = acc[r];
        rs[r] += v;
        const float x  = v * INV_T;
        const float nm = fmaxf(mx[r], x);
        sm[r] = sm[r] * __expf(mx[r] - nm) + __expf(x - nm);
        mx[r] = nm;
        if (n == rowbase + r) dg[r] = v;            // diagonal capture
      }
    }
    __syncthreads();                                // all waves done reading buf before reuse
  }

  // cross-lane reduction within each 16-lane half (xor < 16 keeps halves separate)
  #pragma unroll
  for (int r = 0; r < 8; ++r) {
    float m = mx[r], sme = sm[r], rsum = rs[r], d = dg[r];
    #pragma unroll
    for (int off = 1; off < 16; off <<= 1) {
      const float mo = __shfl_xor(m, off, 32);
      const float so = __shfl_xor(sme, off, 32);
      const float nm = fmaxf(m, mo);
      sme  = sme * __expf(m - nm) + so * __expf(mo - nm);
      m    = nm;
      rsum += __shfl_xor(rsum, off, 32);
      d    += __shfl_xor(d, off, 32);
    }
    if (lr == r) {
      const int row = rowbase + r;
      loss_row[row]   = m + __logf(sme) - INV_T * d;  // logsumexp(S/T) - S_ii/T
      diag_row[row]   = d;
      rowsum_row[row] = rsum;
    }
  }
}

// ---------------- phase 3: deterministic final reduction ----------------
__global__ __launch_bounds__(256) void finalize_kernel(const float* __restrict__ loss_row,
                                                       const float* __restrict__ diag_row,
                                                       const float* __restrict__ rowsum_row,
                                                       float* __restrict__ out) {
  __shared__ float sl[256], sd[256], sr[256];
  const int t = threadIdx.x;
  float a = 0.0f, b = 0.0f, c = 0.0f;
  for (int i = t; i < B_N; i += 256) { a += loss_row[i]; b += diag_row[i]; c += rowsum_row[i]; }
  sl[t] = a; sd[t] = b; sr[t] = c;
  __syncthreads();
  for (int off = 128; off; off >>= 1) {
    if (t < off) { sl[t] += sl[t + off]; sd[t] += sd[t + off]; sr[t] += sr[t + off]; }
    __syncthreads();
  }
  if (t == 0) {
    out[0] = sl[0] / (float)B_N;                                    // mean loss
    out[1] = sd[0] / (float)B_N;                                    // cos_an_pos
    out[2] = (sr[0] - sd[0]) / ((float)B_N * (float)(B_N - 1));     // cos_an_neg
  }
}

// ---------------- launcher ----------------
extern "C" void kernel_launch(void* const* d_in, const int* in_sizes, int n_in,
                              void* d_out, int out_size, void* d_ws, size_t ws_size,
                              hipStream_t stream) {
  (void)in_sizes; (void)n_in; (void)out_size; (void)ws_size;
  const float* anchors = (const float*)d_in[0];
  const float* poss    = (const float*)d_in[1];
  char* ws = (char*)d_ws;

  const size_t MAT = (size_t)B_N * KP * 2;          // 10,616,832 B per bf16 panel
  __bf16* Ahi = (__bf16*)(ws);
  __bf16* Alo = (__bf16*)(ws + MAT);
  __bf16* Phi = (__bf16*)(ws + 2 * MAT);
  __bf16* Plo = (__bf16*)(ws + 3 * MAT);
  float* loss_row   = (float*)(ws + 4 * MAT);
  float* diag_row   = loss_row + B_N;
  float* rowsum_row = diag_row + B_N;

  prep_kernel<<<(2 * B_N) / 8, 256, 0, stream>>>(anchors, poss, Ahi, Alo, Phi, Plo);
  ntxent_main<<<B_N / MBLK, 128, LDS_TOT, stream>>>(Ahi, Alo, Phi, Plo,
                                                    loss_row, diag_row, rowsum_row);
  finalize_kernel<<<1, 256, 0, stream>>>(loss_row, diag_row, rowsum_row, (float*)d_out);
}
